// RNNAggregator_76433238000425
// MI455X (gfx1250) — matrix operand order repaired and checked by reading, not compiled, not run on hardware
//
#include <hip/hip_runtime.h>
#include <cstdint>
#include <cstddef>

// Problem constants (from the reference)
#define NROWS 8192
#define SSTEPS 25
#define DDIM 256
#define UDIM 256
#define KTOT 512      // D + U (concatenated LSTM GEMM K)
#define G4 1024       // 4*U gate columns

// CDNA5 WMMA types
typedef __bf16 bf16_t;
typedef __attribute__((ext_vector_type(16))) __bf16 v16bf;
typedef __attribute__((ext_vector_type(8)))  float  v8f;

// TDM descriptor vector types
typedef __attribute__((ext_vector_type(4))) unsigned int u32x4;
typedef __attribute__((ext_vector_type(4))) int          i32x4;
typedef __attribute__((ext_vector_type(8))) int          i32x8;

#if __has_builtin(__builtin_amdgcn_tensor_load_to_lds)
#define USE_TDM 1
#else
#define USE_TDM 0
#endif

union FragB16 { uint4 q[2]; v16bf v; };

__device__ __forceinline__ unsigned short f2bf(float f) {
  union { float f; unsigned u; } x; x.f = f;
  unsigned r = x.u + 0x7FFFu + ((x.u >> 16) & 1u);   // round-to-nearest-even
  return (unsigned short)(r >> 16);
}
__device__ __forceinline__ unsigned pack2(float a, float b) {
  return (unsigned)f2bf(a) | ((unsigned)f2bf(b) << 16);
}

#if USE_TDM
// ---------------------------------------------------------------------------
// Tensor Data Mover: 2-D tile load global->LDS, bf16 elements (data_size=2B).
// D# group0: count=1 | lds_addr | global_addr(57b) | type=2.
// D# group1 bit-packing per CDNA5 ISA 8.4; optional LDS padding of 4 dwords
// every 32 dwords reproduces the 72-ushort padded A row stride.
// All fields are workgroup-uniform -> land in SGPRs as the ISA requires.
// ---------------------------------------------------------------------------
__device__ __forceinline__ void tdm_load_2d(unsigned lds_off, const void* gptr,
                                            unsigned tile0, unsigned tile1,
                                            unsigned long long stride0,
                                            int padded) {
  const unsigned long long ga = (unsigned long long)(uintptr_t)gptr;
  u32x4 g0;
  g0.x = 1u;                                             // count=1 (user D#)
  g0.y = lds_off;                                        // lds_addr (bytes)
  g0.z = (unsigned)(ga & 0xFFFFFFFFull);                 // global_addr[31:0]
  g0.w = (unsigned)((ga >> 32) & 0x1FFFFFFull)           // global_addr[56:32]
         | (2u << 30);                                   // type = 2 ("image")
  unsigned flags = (1u << 16);                           // data_size = 2 bytes
  if (padded) flags |= (1u << 20)                        // pad_enable
                    | (4u << 22)                         // pad_interval: 32 dw
                    | (3u << 25);                        // pad_amount: 4 dw
  i32x8 g1;
  g1[0] = (int)flags;                                    // [31:0]
  g1[1] = (int)((tile0 & 0xFFFFu) << 16);                // tensor_dim0[15:0] @48
  g1[2] = (int)((tile0 >> 16) | ((tile1 & 0xFFFFu) << 16)); // td0 hi | td1 lo
  g1[3] = (int)((tile1 >> 16) | (tile0 << 16));          // td1 hi | tile_dim0
  g1[4] = (int)(tile1 & 0xFFFFu);                        // tile_dim1 (tile_dim2=0)
  g1[5] = (int)(stride0 & 0xFFFFFFFFull);                // tensor_dim0_stride lo
  g1[6] = (int)((stride0 >> 32) & 0xFFFFull);            // stride hi (dim1_stride=0)
  g1[7] = 0;
  const i32x4 z4 = {0, 0, 0, 0};
#if defined(__clang_major__) && (__clang_major__ >= 23)
  const i32x8 z8 = {0, 0, 0, 0, 0, 0, 0, 0};
  __builtin_amdgcn_tensor_load_to_lds(g0, g1, z4, z4, z8, 0);
#else
  __builtin_amdgcn_tensor_load_to_lds(g0, g1, z4, z4, 0);
#endif
}
#endif  // USE_TDM

// ---------------------------------------------------------------------------
// Weight packing: B-fragment layout for v_wmma_f32_16x16x32_bf16.
// For a (ktile,ntile) 32x16 B tile: lane l holds column N=n0+(l&15), 16 K values in AMD table order.
// Packed index = ((nt*KT + kt)*32 + lane)*16 + e
// ---------------------------------------------------------------------------
__global__ void pack_lstm_kernel(const float* __restrict__ W,   // [256,1024]
                                 const float* __restrict__ Uw,  // [256,1024]
                                 unsigned short* __restrict__ dst) {
  const unsigned g    = blockIdx.x * 256u + threadIdx.x;  // 524288 total
  const unsigned e    = g & 15u;
  const unsigned lane = (g >> 4) & 31u;
  const unsigned kt   = (g >> 9) & 15u;
  const unsigned nt   = g >> 13;                          // 0..63
  const unsigned k    = kt * 32u + ((e & 8u) << 1) + ((lane >> 4) << 3) + (e & 7u); // 0..511
  const unsigned col  = nt * 16u + (lane & 15u);          // 0..1023
  const float s = (k < 256u) ? W[k * 1024u + col] : Uw[(k - 256u) * 1024u + col];
  dst[g] = f2bf(s);
}

__global__ void pack_head_kernel(const float* __restrict__ W,   // [256,256]
                                 unsigned short* __restrict__ dst) {
  const unsigned g    = blockIdx.x * 256u + threadIdx.x;  // 65536 total
  const unsigned e    = g & 15u;
  const unsigned lane = (g >> 4) & 31u;
  const unsigned kt   = (g >> 9) & 7u;
  const unsigned nt   = g >> 12;                          // 0..15
  const unsigned k    = kt * 32u + ((e & 8u) << 1) + ((lane >> 4) << 3) + (e & 7u);
  const unsigned col  = nt * 16u + (lane & 15u);
  dst[g] = f2bf(W[k * 256u + col]);
}

__global__ void init_state_kernel(float* __restrict__ c,
                                  unsigned short* __restrict__ h0) {
  const unsigned i = blockIdx.x * 256u + threadIdx.x;     // N*U = 2M
  c[i] = 0.f;
  h0[i] = 0;
}

// ---------------------------------------------------------------------------
// One LSTM timestep, fused: gates = [x_t | h] @ [W;U] + b ; cell update.
// Workgroup: 256 threads (8 waves). Tile: 128 rows x 16 u-cols x 4 gates.
//   wave w: gate = w&3, row-half = w>>2; 4 sub-tiles of 16 rows each.
// K = 512 in 8 chunks of 64. x_t chunks: VALU f32->bf16 staging; h chunks and
// packed-B chunks: TDM tensor_load_to_lds (pure bf16 copies), waited with
// s_wait_tensorcnt by wave 0 before the workgroup barrier.
// ---------------------------------------------------------------------------
#define MT  128   // rows per workgroup
#define KC  64    // K chunk
#define AST 72    // A LDS row stride (ushorts): 144B rows -> 16B aligned, bank-spread
#define GST 18    // gates LDS row stride (floats): avoids lane-group bank clash

__global__ __launch_bounds__(256)
void lstm_step_kernel(const float* __restrict__ neigh,          // [N,S,D] f32
                      const unsigned short* __restrict__ hin,   // [N,U] bf16
                      unsigned short* __restrict__ hout,        // [N,U] bf16
                      float* __restrict__ cbuf,                 // [N,U] f32
                      const unsigned short* __restrict__ Bpack, // packed [64][16][32][16]
                      const float* __restrict__ bias,           // [4U]
                      int step) {
  __shared__ unsigned short Alds[MT * AST];      // 18,432 B
  __shared__ unsigned short Blds[8 * 32 * 16];   //  8,192 B
  __shared__ float          Glds[4 * MT * GST];  // 36,864 B  (total 63,488 B)

  const int tid  = threadIdx.x;
  const int wave = tid >> 5;
  const int lane = tid & 31;
  const int m0   = blockIdx.x * MT;
  const int ublk = blockIdx.y;            // 0..15 -> u columns [16*ublk, +16)
  const int gate = wave & 3;
  const int half = wave >> 2;

  v8f acc[4];
  const v8f zero = {0.f, 0.f, 0.f, 0.f, 0.f, 0.f, 0.f, 0.f};
#pragma unroll
  for (int st = 0; st < 4; ++st) acc[st] = zero;

  const int r  = tid >> 1;                // 0..127 (A staging row)
  const int cg = (tid & 1) * 32;          // 0 or 32 (A staging col group)

  for (int kc = 0; kc < 8; ++kc) {
    __syncthreads();
    // ---- stage A chunk [128 x 64] bf16 into LDS ----
    if (kc < 4) {  // x_t part from f32 neigh: convert in VALU
      const float* src = neigh + (size_t)(m0 + r) * SSTEPS * DDIM
                               + (size_t)step * DDIM + kc * KC + cg;
      unsigned short* dst = &Alds[r * AST + cg];
#pragma unroll
      for (int j = 0; j < 8; ++j) {
        float4 v = reinterpret_cast<const float4*>(src)[j];
        uint2 p; p.x = pack2(v.x, v.y); p.y = pack2(v.z, v.w);
        reinterpret_cast<uint2*>(dst)[j] = p;
      }
    } else {       // h part, already bf16
#if USE_TDM
      if (wave == 0) {
        // 128 rows x 64 elems, global row stride 256 elems; padded LDS rows
        // (32 dw data + 4 dw pad = 36 dw = 72 ushorts = AST).
        tdm_load_2d((unsigned)(uintptr_t)&Alds[0],
                    hin + (size_t)m0 * UDIM + (kc - 4) * KC,
                    /*tile0=*/KC, /*tile1=*/MT, /*stride0=*/UDIM, /*pad=*/1);
      }
#else
      const unsigned short* src = hin + (size_t)(m0 + r) * UDIM + (kc - 4) * KC + cg;
      unsigned short* dst = &Alds[r * AST + cg];
#pragma unroll
      for (int j = 0; j < 4; ++j)
        reinterpret_cast<uint4*>(dst)[j] = reinterpret_cast<const uint4*>(src)[j];
#endif
    }
    // ---- stage B chunk: 4 gates x 2 ktiles (2KB contiguous per gate) ----
#if USE_TDM
    if (wave == 0) {
      // tile: 1024 elems x 4 gates; stride between gates = 16*16*512 elems.
      tdm_load_2d((unsigned)(uintptr_t)&Blds[0],
                  Bpack + ((size_t)(ublk * 16 + kc * 2)) * 512,
                  /*tile0=*/1024, /*tile1=*/4, /*stride0=*/131072ull, /*pad=*/0);
      __builtin_amdgcn_s_wait_tensorcnt(0);
    }
#else
    {
      const int seg = wave;               // seg = g*2 + ktl
      const int g   = seg >> 1, ktl = seg & 1;
      const int kt  = kc * 2 + ktl;
      const unsigned short* src =
          Bpack + ((size_t)((g * 16 + ublk) * 16 + kt) * 32 + lane) * 16;
      unsigned short* dst = &Blds[(seg * 32 + lane) * 16];
      reinterpret_cast<uint4*>(dst)[0] = reinterpret_cast<const uint4*>(src)[0];
      reinterpret_cast<uint4*>(dst)[1] = reinterpret_cast<const uint4*>(src)[1];
    }
#endif
    __syncthreads();
    // ---- WMMA: 2 ktiles x 4 row sub-tiles ----
#pragma unroll
    for (int ktl = 0; ktl < 2; ++ktl) {
      FragB16 bf;
      const unsigned short* bp = &Blds[((gate * 2 + ktl) * 32 + lane) * 16];
      bf.q[0] = reinterpret_cast<const uint4*>(bp)[0];
      bf.q[1] = reinterpret_cast<const uint4*>(bp)[1];
#pragma unroll
      for (int st = 0; st < 4; ++st) {
        const int row = half * 64 + st * 16 + (lane & 15);
        const int kb  = ktl * 32 + ((lane >> 4) << 3);  // lanes<16: K 0..7/16..23
        FragB16 af;
        const unsigned short* ap = &Alds[row * AST + kb];
        af.q[0] = reinterpret_cast<const uint4*>(ap)[0];
        af.q[1] = reinterpret_cast<const uint4*>(ap + 16)[0];
        acc[st] = __builtin_amdgcn_wmma_f32_16x16x32_bf16(
            false, af.v, false, bf.v, (short)0, acc[st], false, false);
      }
    }
  }
  __syncthreads();
  // ---- accumulators -> gates LDS (C/D layout: M = v + 8*(lane>=16), N = lane&15)
#pragma unroll
  for (int st = 0; st < 4; ++st) {
    const int rbase = half * 64 + st * 16 + ((lane >> 4) << 3);
    const int col   = lane & 15;
#pragma unroll
    for (int v = 0; v < 8; ++v)
      Glds[(gate * MT + rbase + v) * GST + col] = acc[st][v];
  }
  __syncthreads();
  // ---- fused LSTM cell update: 128x16 elements, 8 per thread ----
#pragma unroll
  for (int j = 0; j < 8; ++j) {
    const int e   = tid * 8 + j;
    const int row = e >> 4;
    const int col = e & 15;
    const int n   = m0 + row;
    const int u   = ublk * 16 + col;
    float gi = Glds[(0 * MT + row) * GST + col] + bias[u];
    float gf = Glds[(1 * MT + row) * GST + col] + bias[256 + u];
    float gg = Glds[(2 * MT + row) * GST + col] + bias[512 + u];
    float go = Glds[(3 * MT + row) * GST + col] + bias[768 + u];
    gi = 1.f / (1.f + __expf(-gi));
    gf = 1.f / (1.f + __expf(-gf));
    gg = tanhf(gg);
    go = 1.f / (1.f + __expf(-go));
    const size_t idx = (size_t)n * UDIM + u;
    const float c = gf * cbuf[idx] + gi * gg;
    cbuf[idx] = c;
    hout[idx] = f2bf(go * tanhf(c));
  }
}

// ---------------------------------------------------------------------------
// Head: out_pre[:, 0:256] = self @ W_self ; out_pre[:, 256:512] = h @ W_neigh
// One 16x16 WMMA tile per wave, K=256, B fragments from packed global (L2 hot).
// ---------------------------------------------------------------------------
__global__ __launch_bounds__(256)
void head_gemm_kernel(const float* __restrict__ selfN,          // [N,256] f32
                      const unsigned short* __restrict__ hfin,  // [N,256] bf16
                      const unsigned short* __restrict__ BpS,
                      const unsigned short* __restrict__ BpN,
                      float* __restrict__ outp) {               // [N,512] f32
  const int tid  = threadIdx.x;
  const int wave = tid >> 5;
  const int lane = tid & 31;
  const int gtile   = blockIdx.x * 8 + wave;    // 0..16383
  const int halfsel = gtile >> 13;              // 0: self, 1: neigh
  const int rem     = gtile & 8191;
  const int mt      = rem >> 4;                 // 0..511
  const int nt      = rem & 15;                 // 0..15
  const int mrow  = mt * 16 + (lane & 15);
  const int kbsel = (lane >> 4) << 3;

  const unsigned short* Bp = halfsel ? BpN : BpS;
  v8f acc = {0.f, 0.f, 0.f, 0.f, 0.f, 0.f, 0.f, 0.f};
#pragma unroll
  for (int kt = 0; kt < 8; ++kt) {
    FragB16 af, bf;
    if (halfsel == 0) {
      const float* ap = selfN + (size_t)mrow * 256 + kt * 32 + kbsel;
      float4 v0 = reinterpret_cast<const float4*>(ap)[0];
      float4 v1 = reinterpret_cast<const float4*>(ap)[1];
      float4 v2 = reinterpret_cast<const float4*>(ap + 16)[0];
      float4 v3 = reinterpret_cast<const float4*>(ap + 16)[1];
      af.q[0].x = pack2(v0.x, v0.y); af.q[0].y = pack2(v0.z, v0.w);
      af.q[0].z = pack2(v1.x, v1.y); af.q[0].w = pack2(v1.z, v1.w);
      af.q[1].x = pack2(v2.x, v2.y); af.q[1].y = pack2(v2.z, v2.w);
      af.q[1].z = pack2(v3.x, v3.y); af.q[1].w = pack2(v3.z, v3.w);
    } else {
      const unsigned short* ap = hfin + (size_t)mrow * 256 + kt * 32 + kbsel;
      af.q[0] = reinterpret_cast<const uint4*>(ap)[0];
      af.q[1] = reinterpret_cast<const uint4*>(ap + 16)[0];
    }
    const unsigned short* bp = Bp + ((size_t)(nt * 8 + kt) * 32 + lane) * 16;
    bf.q[0] = reinterpret_cast<const uint4*>(bp)[0];
    bf.q[1] = reinterpret_cast<const uint4*>(bp)[1];
    acc = __builtin_amdgcn_wmma_f32_16x16x32_bf16(
        false, af.v, false, bf.v, (short)0, acc, false, false);
  }
  const int col   = halfsel * 256 + nt * 16 + (lane & 15);
  const int rbase = mt * 16 + ((lane >> 4) << 3);
#pragma unroll
  for (int v = 0; v < 8; ++v)
    outp[(size_t)(rbase + v) * 512 + col] = acc[v];
}

// ---------------------------------------------------------------------------
// BatchNorm (training stats) + ReLU over out_pre [8192, 512], in place.
// ---------------------------------------------------------------------------
__global__ __launch_bounds__(256)
void bn_stats_kernel(const float* __restrict__ x, float* __restrict__ stats) {
  __shared__ float s1[256], s2[256];
  const int c = blockIdx.x;  // column 0..511
  float sum = 0.f, sq = 0.f;
  for (int r = threadIdx.x; r < NROWS; r += 256) {
    const float v = x[(size_t)r * 512 + c];
    sum += v; sq += v * v;
  }
  s1[threadIdx.x] = sum; s2[threadIdx.x] = sq;
  __syncthreads();
  for (int o = 128; o > 0; o >>= 1) {
    if (threadIdx.x < o) {
      s1[threadIdx.x] += s1[threadIdx.x + o];
      s2[threadIdx.x] += s2[threadIdx.x + o];
    }
    __syncthreads();
  }
  if (threadIdx.x == 0) {
    const float mean = s1[0] / (float)NROWS;
    const float var  = s2[0] / (float)NROWS - mean * mean;
    stats[c]       = mean;
    stats[512 + c] = rsqrtf(var + 1e-3f);
  }
}

__global__ void bn_apply_kernel(float* __restrict__ x,
                                const float* __restrict__ stats,
                                const float* __restrict__ gamma,
                                const float* __restrict__ beta) {
  const unsigned i = blockIdx.x * 256u + threadIdx.x;  // 8192*512
  const unsigned c = i & 511u;
  const float v = (x[i] - stats[c]) * stats[512 + c] * gamma[c] + beta[c];
  x[i] = v > 0.f ? v : 0.f;
}

// ---------------------------------------------------------------------------
extern "C" void kernel_launch(void* const* d_in, const int* in_sizes, int n_in,
                              void* d_out, int out_size, void* d_ws, size_t ws_size,
                              hipStream_t stream) {
  (void)in_sizes; (void)n_in; (void)out_size; (void)ws_size;
  const float* selfN = (const float*)d_in[0];
  const float* neigh = (const float*)d_in[1];
  // d_in[2] len_adj_nodes: unused by the reference
  const float* Wl    = (const float*)d_in[3];
  const float* Ul    = (const float*)d_in[4];
  const float* bl    = (const float*)d_in[5];
  const float* Ws    = (const float*)d_in[6];
  const float* Wn    = (const float*)d_in[7];
  const float* gamma = (const float*)d_in[8];
  const float* beta  = (const float*)d_in[9];
  float* out = (float*)d_out;

  char* ws = (char*)d_ws;
  unsigned short* BpL = (unsigned short*)(ws + 0);         // 1,048,576 B
  unsigned short* BpS = (unsigned short*)(ws + 1048576);   //   131,072 B
  unsigned short* BpN = (unsigned short*)(ws + 1179648);   //   131,072 B
  unsigned short* hA  = (unsigned short*)(ws + 1310720);   // 4,194,304 B
  unsigned short* hB  = (unsigned short*)(ws + 5505024);   // 4,194,304 B
  float*          cb  = (float*)(ws + 9699328);            // 8,388,608 B
  float*          st  = (float*)(ws + 18087936);           //     4,096 B

  // Pack weights to bf16 WMMA fragment order; zero h0/c0 (deterministic per call).
  pack_lstm_kernel<<<2048, 256, 0, stream>>>(Wl, Ul, BpL);
  pack_head_kernel<<<256, 256, 0, stream>>>(Ws, BpS);
  pack_head_kernel<<<256, 256, 0, stream>>>(Wn, BpN);
  init_state_kernel<<<8192, 256, 0, stream>>>(cb, hA);

  // 25 sequential LSTM steps, h ping-pong to avoid cross-workgroup RAW races.
  dim3 sgrid(NROWS / MT, UDIM / 16);
  for (int t = 0; t < SSTEPS; ++t) {
    const unsigned short* hin = (t & 1) ? hB : hA;
    unsigned short*      hout = (t & 1) ? hA : hB;
    lstm_step_kernel<<<sgrid, 256, 0, stream>>>(neigh, hin, hout, cb, BpL, bl, t);
  }
  const unsigned short* hfin = hB;  // t=24 (even) wrote hB

  head_gemm_kernel<<<2048, 256, 0, stream>>>(selfN, hfin, BpS, BpN, out);
  bn_stats_kernel<<<512, 256, 0, stream>>>(out, st);
  bn_apply_kernel<<<(NROWS * 512) / 256, 256, 0, stream>>>(out, st, gamma, beta);
}
